// LocalRetro_58926951301831
// MI455X (gfx1250) — matrix operand order repaired and checked by reading, not compiled, not run on hardware
//
#include <hip/hip_runtime.h>

typedef _Float16 v16h __attribute__((ext_vector_type(16)));
typedef _Float16 v8h  __attribute__((ext_vector_type(8)));
typedef float    v8f  __attribute__((ext_vector_type(8)));
typedef float    v4f  __attribute__((ext_vector_type(4)));

#define NODE_IN 74
#define EDGE_IN 13
#define OUTD    64
#define EHID    128
#define NSTEPS  6
#define OO      4096   // OUTD*OUTD
#define LN_EPS  1e-5f

// ---------------------------------------------------------------- node proj
__global__ __launch_bounds__(256)
void node_proj_kernel(const float* __restrict__ nf, const float* __restrict__ Wp,
                      const float* __restrict__ bp, float* __restrict__ h, int V) {
    int idx = blockIdx.x * 256 + threadIdx.x;
    if (idx >= V * OUTD) return;
    int v = idx >> 6, o = idx & 63;
    const float* row = nf + (size_t)v * NODE_IN;
    float acc = bp[o];
    #pragma unroll
    for (int j = 0; j < NODE_IN; ++j) acc += row[j] * Wp[j * OUTD + o];
    h[idx] = fmaxf(acc, 0.f);
}

// ---------------------------------------------------------------- edge hidden (f16 out, relu'd)
__global__ __launch_bounds__(256)
void edge_hidden_kernel(const float* __restrict__ ef, const float* __restrict__ We1,
                        const float* __restrict__ be1, _Float16* __restrict__ A, int E) {
    int idx = blockIdx.x * 256 + threadIdx.x;
    if (idx >= E * EHID) return;
    int e = idx >> 7, k = idx & 127;
    const float* row = ef + (size_t)e * EDGE_IN;
    float acc = be1[k];
    #pragma unroll
    for (int j = 0; j < EDGE_IN; ++j) acc += row[j] * We1[j * EHID + k];
    A[idx] = (_Float16)fmaxf(acc, 0.f);
}

// ---------------------------------------------------------------- We2 (128x4096 f32) -> BT (4096x128 f16)
__global__ __launch_bounds__(256)
void transpose_f16_kernel(const float* __restrict__ in, _Float16* __restrict__ out) {
    int idx = blockIdx.x * 256 + threadIdx.x;     // idx = n*128 + k (coalesced writes)
    if (idx >= EHID * OO) return;
    int n = idx >> 7, k = idx & 127;
    out[idx] = (_Float16)in[(size_t)k * OO + n];
}

// ---------------------------------------------------------------- zero
__global__ __launch_bounds__(256)
void zero_kernel(float* __restrict__ p, int n) {
    int idx = blockIdx.x * 256 + threadIdx.x;
    if (idx < n) p[idx] = 0.f;
}

// ---------------------------------------------------------------- big WMMA GEMM:
//   C[e, n] = sum_k A[e,k] * BT[n,k] + be2[n]
//   A: E x 128 (f16, row-major), BT: 4096 x 128 (f16, row-major = We2^T)
//   one wave -> 16x64 strip (4 N-tiles), K = 4 x 32 via v_wmma_f32_16x16x32_f16
__global__ __launch_bounds__(256)
void edge_gemm_kernel(const _Float16* __restrict__ A, const _Float16* __restrict__ BT,
                      const float* __restrict__ be2, _Float16* __restrict__ C, int E) {
    const int NTG = (OO / 16) / 4;                // 64 groups of 4 n-tiles
    int wave = (blockIdx.x * 256 + threadIdx.x) >> 5;
    int lane = threadIdx.x & 31;
    int mtiles = (E + 15) >> 4;
    if (wave >= mtiles * NTG) return;             // wave-uniform
    int mt = wave / NTG;
    int ng = wave - mt * NTG;

    int halfsel = lane >> 4;                      // 0: lanes 0-15, 1: lanes 16-31
    int lrow    = lane & 15;
    int row     = mt * 16 + lrow;                 // A row for this lane's fragment
    bool rowok  = row < E;

    // A fragments: lane-half picks K octets {base..base+7, base+16..base+23}
    v16h af[4];
    #pragma unroll
    for (int kb = 0; kb < 4; ++kb) {
        if (rowok) {
            const _Float16* ap = A + (size_t)row * EHID + kb * 32 + halfsel * 8;
            v8h lo = *(const v8h*)(ap);
            v8h hi = *(const v8h*)(ap + 16);
            af[kb] = __builtin_shufflevector(lo, hi, 0,1,2,3,4,5,6,7,8,9,10,11,12,13,14,15);
        } else {
            af[kb] = (v16h)(_Float16)0.f;
        }
    }

    v8f c[4] = {v8f{}, v8f{}, v8f{}, v8f{}};
    #pragma unroll
    for (int kb = 0; kb < 4; ++kb) {
        v16h bf[4];
        #pragma unroll
        for (int t = 0; t < 4; ++t) {
            // B fragment: lane = column, lane-half picks K octets {kb*32 + half*8 .. +7, +16 .. +23}
            const _Float16* bp = BT + (size_t)((ng * 4 + t) * 16 + lrow) * EHID
                                    + kb * 32 + halfsel * 8;
            v8h lo = *(const v8h*)(bp);
            v8h hi = *(const v8h*)(bp + 16);
            bf[t] = __builtin_shufflevector(lo, hi, 0,1,2,3,4,5,6,7,8,9,10,11,12,13,14,15);
        }
        #pragma unroll
        for (int t = 0; t < 4; ++t) {
            c[t] = __builtin_amdgcn_wmma_f32_16x16x32_f16(false, af[kb], false, bf[t],
                                                          (short)0, c[t], false, false);
        }
    }

    int mof = halfsel * 8;
    #pragma unroll
    for (int t = 0; t < 4; ++t) {
        int col = (ng * 4 + t) * 16 + lrow;
        float bias = be2[col];
        #pragma unroll
        for (int r = 0; r < 8; ++r) {
            int m = mt * 16 + r + mof;
            if (m < E) {
                _Float16 val = (_Float16)(c[t][r] + bias);
                __builtin_nontemporal_store(val, &C[(size_t)m * OO + col]);
            }
        }
    }
}

// ---------------------------------------------------------------- message pass scatter:
//   msg[e,:] = h[src[e],:] (1x64) * W_e (64x64, f16);  agg[dst[e],:] += msg
//   8 threads/edge, 8 outputs each via 16B NT loads; h[src] staged in LDS
__global__ __launch_bounds__(256)
void mp_scatter_kernel(const float* __restrict__ h, const _Float16* __restrict__ W,
                       const int* __restrict__ src, const int* __restrict__ dst,
                       float* __restrict__ agg, int E) {
    __shared__ float hs[32 * OUTD];
    int tid = threadIdx.x;
    int eb  = tid >> 3;                 // edge slot in block (0..31)
    int t   = tid & 7;                  // thread in edge
    int e   = blockIdx.x * 32 + eb;
    bool ok = e < E;
    int s = ok ? src[e] : 0;
    if (ok) {
        *(v4f*)(hs + eb * OUTD + t * 8)     = *(const v4f*)(h + (size_t)s * OUTD + t * 8);
        *(v4f*)(hs + eb * OUTD + t * 8 + 4) = *(const v4f*)(h + (size_t)s * OUTD + t * 8 + 4);
    }
    __syncthreads();
    if (!ok) return;

    float a[8] = {0.f, 0.f, 0.f, 0.f, 0.f, 0.f, 0.f, 0.f};
    const _Float16* wp = W + (size_t)e * OO + t * 8;
    const float* hrow  = hs + eb * OUTD;
    #pragma unroll 8
    for (int i = 0; i < OUTD; ++i) {
        float hv = hrow[i];
        v8h w = __builtin_nontemporal_load((const v8h*)(wp + (size_t)i * OUTD));
        #pragma unroll
        for (int j = 0; j < 8; ++j) a[j] += (float)w[j] * hv;   // v_fma_mix_f32
    }
    int d = dst[e];
    float* ag = agg + (size_t)d * OUTD + t * 8;
    #pragma unroll
    for (int j = 0; j < 8; ++j) atomicAdd(ag + j, a[j]);
}

// ---------------------------------------------------------------- h = relu(agg + bconv); agg = 0
__global__ __launch_bounds__(256)
void relu_bias_kernel(float* __restrict__ agg, const float* __restrict__ bconv,
                      float* __restrict__ h, int V) {
    int idx = blockIdx.x * 256 + threadIdx.x;
    if (idx >= V * OUTD) return;
    int o = idx & 63;
    h[idx] = fmaxf(agg[idx] + bconv[o], 0.f);
    agg[idx] = 0.f;
}

// ---------------------------------------------------------------- bond_feats = [h_src, h_dst] @ WB + bB
__global__ __launch_bounds__(256)
void bond_kernel(const float* __restrict__ h, const int* __restrict__ src,
                 const int* __restrict__ dst, const float* __restrict__ WB,
                 const float* __restrict__ bB, float* __restrict__ bond, int E) {
    int idx = blockIdx.x * 256 + threadIdx.x;
    if (idx >= E * OUTD) return;
    int e = idx >> 6, o = idx & 63;
    const float* hsrow = h + (size_t)src[e] * OUTD;
    const float* hdrow = h + (size_t)dst[e] * OUTD;
    float acc = bB[o];
    #pragma unroll
    for (int i = 0; i < OUTD; ++i) {
        acc += hsrow[i] * WB[i * OUTD + o];
        acc += hdrow[i] * WB[(OUTD + i) * OUTD + o];
    }
    bond[idx] = acc;
}

// ---------------------------------------------------------------- LayerNorm over 64-wide rows; 1 wave32/row
__global__ __launch_bounds__(256)
void ln_kernel(const float* __restrict__ x, const float* __restrict__ gamma,
               const float* __restrict__ beta, float* __restrict__ out, int rows) {
    int row = blockIdx.x * 8 + (threadIdx.x >> 5);
    if (row >= rows) return;                     // wave-uniform
    int l = threadIdx.x & 31;
    float x0 = x[(size_t)row * OUTD + l];
    float x1 = x[(size_t)row * OUTD + 32 + l];
    float s = x0 + x1;
    #pragma unroll
    for (int m = 16; m >= 1; m >>= 1) s += __shfl_xor(s, m, 32);
    float mu = s * (1.f / 64.f);
    float d0 = x0 - mu, d1 = x1 - mu;
    float v = d0 * d0 + d1 * d1;
    #pragma unroll
    for (int m = 16; m >= 1; m >>= 1) v += __shfl_xor(v, m, 32);
    float inv = rsqrtf(v * (1.f / 64.f) + LN_EPS);
    out[(size_t)row * OUTD + l]      = d0 * inv * gamma[l]      + beta[l];
    out[(size_t)row * OUTD + 32 + l] = d1 * inv * gamma[32 + l] + beta[32 + l];
}

// ================================================================ launch
extern "C" void kernel_launch(void* const* d_in, const int* in_sizes, int n_in,
                              void* d_out, int out_size, void* d_ws, size_t ws_size,
                              hipStream_t stream) {
    const float* node_feats = (const float*)d_in[0];
    const float* edge_feats = (const float*)d_in[1];
    const int*   src        = (const int*)  d_in[2];
    const int*   dst        = (const int*)  d_in[3];
    const float* Wp         = (const float*)d_in[4];
    const float* bp         = (const float*)d_in[5];
    const float* We1        = (const float*)d_in[6];
    const float* be1        = (const float*)d_in[7];
    const float* We2        = (const float*)d_in[8];
    const float* be2        = (const float*)d_in[9];
    const float* bconv      = (const float*)d_in[10];
    const float* WB         = (const float*)d_in[11];
    const float* bB         = (const float*)d_in[12];
    const float* gamma      = (const float*)d_in[13];
    const float* beta       = (const float*)d_in[14];

    const int V = in_sizes[0] / NODE_IN;
    const int E = in_sizes[1] / EDGE_IN;

    // workspace partition (256B aligned)
    char* w = (char*)d_ws;
    auto take = [&](size_t bytes) { char* p = w; w += (bytes + 255) & ~(size_t)255; return p; };
    float*     h    = (float*)    take((size_t)V * OUTD * sizeof(float));
    _Float16*  A    = (_Float16*) take((size_t)E * EHID * sizeof(_Float16));
    _Float16*  BT   = (_Float16*) take((size_t)EHID * OO * sizeof(_Float16));
    _Float16*  Wem  = (_Float16*) take((size_t)E * OO * sizeof(_Float16));
    float*     agg  = (float*)    take((size_t)V * OUTD * sizeof(float));
    float*     bond = (float*)    take((size_t)E * OUTD * sizeof(float));

    float* atom_out = (float*)d_out;                       // V*64
    float* bond_out = (float*)d_out + (size_t)V * OUTD;    // E*64

    auto cdiv = [](long long a, long long b) { return (int)((a + b - 1) / b); };

    // 1. h0 = relu(node_feats @ Wp + bp)
    node_proj_kernel<<<cdiv((long long)V * OUTD, 256), 256, 0, stream>>>(node_feats, Wp, bp, h, V);
    // 2. A = relu(edge_feats @ We1 + be1) -> f16
    edge_hidden_kernel<<<cdiv((long long)E * EHID, 256), 256, 0, stream>>>(edge_feats, We1, be1, A, E);
    // 3. BT = We2^T -> f16 (contiguous-K fragments for WMMA B)
    transpose_f16_kernel<<<cdiv(EHID * OO, 256), 256, 0, stream>>>(We2, BT);
    // 4. W_e = A @ We2 + be2  (WMMA, f16 out, NT stores)
    {
        long long waves = (long long)((E + 15) / 16) * ((OO / 16) / 4);
        edge_gemm_kernel<<<cdiv(waves, 8), 256, 0, stream>>>(A, BT, be2, Wem, E);
    }
    // 5. agg = 0
    zero_kernel<<<cdiv((long long)V * OUTD, 256), 256, 0, stream>>>(agg, V * OUTD);
    // 6. message passing steps
    for (int step = 0; step < NSTEPS; ++step) {
        mp_scatter_kernel<<<cdiv(E, 32), 256, 0, stream>>>(h, Wem, src, dst, agg, E);
        relu_bias_kernel<<<cdiv((long long)V * OUTD, 256), 256, 0, stream>>>(agg, bconv, h, V);
    }
    // 7. bond feats
    bond_kernel<<<cdiv((long long)E * OUTD, 256), 256, 0, stream>>>(h, src, dst, WB, bB, bond, E);
    // 8. layer norms -> outputs
    ln_kernel<<<cdiv(V, 8), 256, 0, stream>>>(h, gamma, beta, atom_out, V);
    ln_kernel<<<cdiv(E, 8), 256, 0, stream>>>(bond, gamma, beta, bond_out, E);
}
